// TransfomerAttentionLayer_fusion_50594714747331
// MI455X (gfx1250) — compile-verified, matrix operand order checked by reading, not run on hardware
//
#include <hip/hip_runtime.h>
#include <math.h>

// ---------------- problem constants ----------------
#define NUM_DSTN  60000
#define NUM_EDG   600000
#define D_NODE    172
#define D_EDGE    172
#define D_TIME    100
#define D_OUTC    100
#define D_HEAD    50          // D_OUT / NUM_HEAD
#define KV_IN     444         // D_NODE + D_EDGE + D_TIME
#define KV_KP     448         // padded to multiple of 32
#define Q_IN      272         // D_NODE + D_TIME
#define Q_KP      288
#define O_IN      272         // D_OUT + D_NODE
#define O_KP      288
#define NP        112         // padded output cols (7 tiles of 16)
#define NT        7

// LDS strides (bf16 elements), skewed to avoid bank conflicts
#define KVASTR    456
#define QASTR     296
#define OASTR     296

typedef __attribute__((ext_vector_type(16))) __bf16 v16bf;
typedef __attribute__((ext_vector_type(8)))  float  v8f;

// ---------------- workspace layout (bytes) ----------------
// all offsets 256B aligned
static constexpr size_t BK_OFF = 0;                        // 448*112*2 = 100352
static constexpr size_t BV_OFF = BK_OFF + 100352;
static constexpr size_t BQ_OFF = BV_OFF + 100352;          // 288*112*2 = 64512
static constexpr size_t BO_OFF = BQ_OFF + 64512;
static constexpr size_t Q_OFF  = BO_OFF + 64512;           // 60000*100*4
static constexpr size_t K_OFF  = Q_OFF  + (size_t)NUM_DSTN * D_OUTC * 4;
static constexpr size_t V_OFF  = K_OFF  + (size_t)NUM_EDG  * D_OUTC * 4;
static constexpr size_t SC_OFF = V_OFF  + (size_t)NUM_EDG  * D_OUTC * 4;
static constexpr size_t AG_OFF = SC_OFF + (size_t)NUM_EDG  * 2 * 4;

// ---------------- WMMA fragment loaders ----------------
// A (16x32 bf16, MxK): lane = Khalf*16 + M ; elems 0..7 -> K = Khalf*8 + e,
// elems 8..15 -> K = 16 + Khalf*8 + (e-8)   (ISA 7.12.2)
__device__ __forceinline__ v16bf load_a_frag(const __bf16* sA, int stride,
                                             int kbase, int lane) {
  int m  = lane & 15;
  int kh = (lane >> 4) * 8;
  const __bf16* p = sA + m * stride + kbase + kh;
  v16bf a;
#pragma unroll
  for (int e = 0; e < 8; ++e) a[e] = p[e];          // -> ds_load_b128
#pragma unroll
  for (int e = 0; e < 8; ++e) a[8 + e] = p[16 + e]; // -> ds_load_b128
  return a;
}

// B pre-packed in fragment order: frag[(kt*NP + col)*2 + khalf] is the
// contiguous 32-byte v16bf a lane needs -> two global_load_b128 per fragment.
__device__ __forceinline__ v16bf load_b_frag(const __bf16* B, int kt,
                                             int col, int lane) {
  int kh = (lane >> 4) & 1;
  const v16bf* p = (const v16bf*)B;
  return p[(kt * NP + col) * 2 + kh];
}

// ---------------- kernel 1: pack weights to WMMA-fragment-order bf16 ----------------
__device__ __forceinline__ void pack_one(__bf16* B, const float* __restrict__ w,
                                         int realK, int t) {
  int e   = t & 15;
  int kh  = (t >> 4) & 1;
  int col = (t >> 5) % NP;
  int kt  = (t >> 5) / NP;
  int K   = kt * 32 + kh * 8 + (e < 8 ? e : e + 8);
  float v = (K < realK && col < D_OUTC) ? w[K * D_OUTC + col] : 0.f;
  B[t] = (__bf16)v;
}

__global__ void prep_b_kernel(const float* __restrict__ wq,
                              const float* __restrict__ wk,
                              const float* __restrict__ wv,
                              const float* __restrict__ wo,
                              __bf16* Bq, __bf16* Bk, __bf16* Bv, __bf16* Bo) {
  int t = blockIdx.x * blockDim.x + threadIdx.x;
  if (t < KV_KP * NP) {
    pack_one(Bk, wk, KV_IN, t);
    pack_one(Bv, wv, KV_IN, t);
  }
  if (t < Q_KP * NP) {
    pack_one(Bq, wq, Q_IN, t);
    pack_one(Bo, wo, O_IN, t);
  }
}

// ---------------- kernel 2: Q = [dst_h | cos(time_b)] @ Wq + bq ----------------
__global__ __launch_bounds__(256) void q_gemm_kernel(
    const float* __restrict__ node_feats, const float* __restrict__ time_b,
    const __bf16* __restrict__ Bq, const float* __restrict__ wq_b,
    float* __restrict__ Qbuf) {
  __shared__ __bf16 sA[16 * QASTR];
  int mTile = blockIdx.x;
  int tid = threadIdx.x;
  for (int idx = tid; idx < 16 * Q_KP; idx += 256) {
    int r = idx / Q_KP, c = idx % Q_KP;
    int dst = mTile * 16 + r;
    float v = 0.f;
    if (c < D_NODE)      v = node_feats[(size_t)dst * D_NODE + c];
    else if (c < Q_IN)   v = cosf(time_b[c - D_NODE]);
    sA[r * QASTR + c] = (__bf16)v;
  }
  __syncthreads();
  int wid = tid >> 5, lane = tid & 31;
  if (wid < NT) {                               // wave-uniform: EXEC all ones
    int col = wid * 16 + (lane & 15);
    v8f acc = {};
    for (int kb = 0; kb < Q_KP; kb += 32) {
      v16bf a = load_a_frag(sA, QASTR, kb, lane);
      v16bf b = load_b_frag(Bq, kb >> 5, col, lane);
      acc = __builtin_amdgcn_wmma_f32_16x16x32_bf16(false, a, false, b,
                                                    (short)0, acc, false, false);
    }
    if (col < D_OUTC) {
      float bias = wq_b[col];
      int kh = lane >> 4;
#pragma unroll
      for (int r = 0; r < 8; ++r) {
        size_t row = (size_t)(mTile * 16 + r + 8 * kh);
        Qbuf[row * D_OUTC + col] = acc[r] + bias;
      }
    }
  }
}

// ---------------- kernel 3: K,V = [src_h|edge|cos(dt*w+b)] @ {Wk,Wv} ----------------
__global__ __launch_bounds__(256) void kv_gemm_kernel(
    const float* __restrict__ node_feats, const float* __restrict__ edge_feats,
    const float* __restrict__ edge_dt, const float* __restrict__ time_w,
    const float* __restrict__ time_b, const __bf16* __restrict__ Bk,
    const __bf16* __restrict__ Bv, const float* __restrict__ wk_b,
    const float* __restrict__ wv_b, float* __restrict__ Kbuf,
    float* __restrict__ Vbuf) {
  __shared__ __bf16 sA[16 * KVASTR];
  int mTile = blockIdx.x;
  int tid = threadIdx.x;
  for (int idx = tid; idx < 16 * KV_KP; idx += 256) {
    int r = idx / KV_KP, c = idx % KV_KP;
    int e = mTile * 16 + r;
    float v = 0.f;
    if (c < D_NODE)                 v = node_feats[(size_t)(NUM_DSTN + e) * D_NODE + c];
    else if (c < D_NODE + D_EDGE)   v = edge_feats[(size_t)e * D_EDGE + (c - D_NODE)];
    else if (c < KV_IN) {
      int t = c - (D_NODE + D_EDGE);
      v = cosf(edge_dt[e] * time_w[t] + time_b[t]);
    }
    sA[r * KVASTR + c] = (__bf16)v;
  }
  __syncthreads();
  int wid = tid >> 5, lane = tid & 31;
  if (wid < NT) {
    int col = wid * 16 + (lane & 15);
    v8f ak = {}, av = {};
    for (int kb = 0; kb < KV_KP; kb += 32) {
      v16bf a  = load_a_frag(sA, KVASTR, kb, lane);   // shared by K and V
      v16bf bk = load_b_frag(Bk, kb >> 5, col, lane);
      v16bf bv = load_b_frag(Bv, kb >> 5, col, lane);
      ak = __builtin_amdgcn_wmma_f32_16x16x32_bf16(false, a, false, bk,
                                                   (short)0, ak, false, false);
      av = __builtin_amdgcn_wmma_f32_16x16x32_bf16(false, a, false, bv,
                                                   (short)0, av, false, false);
    }
    if (col < D_OUTC) {
      float bk_ = wk_b[col], bv_ = wv_b[col];
      int kh = lane >> 4;
#pragma unroll
      for (int r = 0; r < 8; ++r) {
        size_t row = (size_t)(mTile * 16 + r + 8 * kh);
        Kbuf[row * D_OUTC + col] = ak[r] + bk_;
        Vbuf[row * D_OUTC + col] = av[r] + bv_;
      }
    }
  }
}

// ---------------- kernel 4: per-edge per-head score + leaky relu ----------------
__global__ void score_kernel(const float* __restrict__ Qbuf,
                             const float* __restrict__ Kbuf,
                             const int* __restrict__ edge_dst,
                             float* __restrict__ score) {
  int e = blockIdx.x * blockDim.x + threadIdx.x;
  if (e >= NUM_EDG) return;
  // rows are 400B and ws offsets are 256B-aligned -> float4 aligned
  const float4* q4 = (const float4*)(Qbuf + (size_t)edge_dst[e] * D_OUTC);
  const float4* k4 = (const float4*)(Kbuf + (size_t)e * D_OUTC);
  float s0 = 0.f, s1 = 0.f;
#pragma unroll
  for (int j = 0; j < 12; ++j) {                       // cols 0..47 -> head 0
    float4 a = q4[j], b = k4[j];
    s0 = fmaf(a.x, b.x, s0); s0 = fmaf(a.y, b.y, s0);
    s0 = fmaf(a.z, b.z, s0); s0 = fmaf(a.w, b.w, s0);
  }
  {
    float4 a = q4[12], b = k4[12];                     // cols 48,49 | 50,51
    s0 = fmaf(a.x, b.x, s0); s0 = fmaf(a.y, b.y, s0);
    s1 = fmaf(a.z, b.z, s1); s1 = fmaf(a.w, b.w, s1);
  }
#pragma unroll
  for (int j = 13; j < 25; ++j) {                      // cols 52..99 -> head 1
    float4 a = q4[j], b = k4[j];
    s1 = fmaf(a.x, b.x, s1); s1 = fmaf(a.y, b.y, s1);
    s1 = fmaf(a.z, b.z, s1); s1 = fmaf(a.w, b.w, s1);
  }
  float2* sc = (float2*)score;
  float2 out;
  out.x = s0 > 0.f ? s0 : 0.2f * s0;
  out.y = s1 > 0.f ? s1 : 0.2f * s1;
  sc[e] = out;
}

// ---------------- kernel 5: segment softmax + weighted V aggregation ----------------
__device__ __forceinline__ int lb_int(const int* a, int n, int v) {
  int lo = 0, hi = n;
  while (lo < hi) { int m = (lo + hi) >> 1; if (a[m] < v) lo = m + 1; else hi = m; }
  return lo;
}

__global__ __launch_bounds__(256) void agg_kernel(
    const float* __restrict__ Vbuf, const float* __restrict__ score,
    const int* __restrict__ edge_dst, float* __restrict__ agg) {
  int d = blockIdx.x * 8 + (threadIdx.x >> 5);   // one wave per destination
  if (d >= NUM_DSTN) return;                     // wave-uniform
  int lane = threadIdx.x & 31;
  int b  = lb_int(edge_dst, NUM_EDG, d);
  int e2 = lb_int(edge_dst, NUM_EDG, d + 1);
  const float2* sc = (const float2*)score;

  float m0 = -3.4e38f, m1 = -3.4e38f;
  for (int e = b + lane; e < e2; e += 32) {
    float2 s = sc[e];
    m0 = fmaxf(m0, s.x);
    m1 = fmaxf(m1, s.y);
  }
#pragma unroll
  for (int off = 16; off; off >>= 1) {
    m0 = fmaxf(m0, __shfl_xor(m0, off, 32));
    m1 = fmaxf(m1, __shfl_xor(m1, off, 32));
  }
  float s0 = 0.f, s1 = 0.f;
  for (int e = b + lane; e < e2; e += 32) {
    float2 s = sc[e];
    s0 += expf(s.x - m0);
    s1 += expf(s.y - m1);
  }
#pragma unroll
  for (int off = 16; off; off >>= 1) {
    s0 += __shfl_xor(s0, off, 32);
    s1 += __shfl_xor(s1, off, 32);
  }
  float r0 = s0 > 0.f ? 1.f / s0 : 0.f;
  float r1 = s1 > 0.f ? 1.f / s1 : 0.f;

  // lanes own columns: c, c+32, c+64, c+96 (head = col/50)
  int c0 = lane, c1 = lane + 32, c2 = lane + 64, c3 = lane + 96;
  float a0 = 0.f, a1 = 0.f, a2 = 0.f, a3 = 0.f;
  for (int e = b; e < e2; ++e) {
    float2 s = sc[e];
    float w0 = expf(s.x - m0) * r0;
    float w1 = expf(s.y - m1) * r1;
    const float* v = Vbuf + (size_t)e * D_OUTC;
    a0 = fmaf(w0, v[c0], a0);                        // cols 0..31  -> head 0
    a1 = fmaf((c1 < D_HEAD ? w0 : w1), v[c1], a1);   // cols 32..63 -> mixed
    a2 = fmaf(w1, v[c2], a2);                        // cols 64..95 -> head 1
    if (c3 < D_OUTC) a3 = fmaf(w1, v[c3], a3);       // cols 96..99
  }
  float* o = agg + (size_t)d * D_OUTC;
  o[c0] = a0; o[c1] = a1; o[c2] = a2;
  if (c3 < D_OUTC) o[c3] = a3;
}

// ---------------- kernel 6: output GEMM + ReLU + LayerNorm ----------------
__global__ __launch_bounds__(256) void out_gemm_ln_kernel(
    const float* __restrict__ agg, const float* __restrict__ node_feats,
    const __bf16* __restrict__ Bo, const float* __restrict__ wo_b,
    const float* __restrict__ ln_g, const float* __restrict__ ln_b,
    float* __restrict__ out) {
  __shared__ __bf16 sA[16 * OASTR];
  __shared__ float  sC[16 * 113];
  int mTile = blockIdx.x, tid = threadIdx.x;
  for (int idx = tid; idx < 16 * O_KP; idx += 256) {
    int r = idx / O_KP, c = idx % O_KP;
    int dst = mTile * 16 + r;
    float v = 0.f;
    if (c < D_OUTC)      v = agg[(size_t)dst * D_OUTC + c];
    else if (c < O_IN)   v = node_feats[(size_t)dst * D_NODE + (c - D_OUTC)];
    sA[r * OASTR + c] = (__bf16)v;
  }
  __syncthreads();
  int wid = tid >> 5, lane = tid & 31;
  if (wid < NT) {
    int col = wid * 16 + (lane & 15);
    v8f acc = {};
    for (int kb = 0; kb < O_KP; kb += 32) {
      v16bf a   = load_a_frag(sA, OASTR, kb, lane);
      v16bf bfr = load_b_frag(Bo, kb >> 5, col, lane);
      acc = __builtin_amdgcn_wmma_f32_16x16x32_bf16(false, a, false, bfr,
                                                    (short)0, acc, false, false);
    }
    if (col < D_OUTC) {
      float bias = wo_b[col];
      int kh = lane >> 4;
#pragma unroll
      for (int r = 0; r < 8; ++r) {
        float v = acc[r] + bias;
        sC[(r + 8 * kh) * 113 + col] = v > 0.f ? v : 0.f;   // ReLU
      }
    }
  }
  __syncthreads();
  if (tid < 16) {                      // per-row LayerNorm
    const float* rp = sC + tid * 113;
    float mean = 0.f;
    for (int c = 0; c < D_OUTC; ++c) mean += rp[c];
    mean *= (1.f / D_OUTC);
    float var = 0.f;
    for (int c = 0; c < D_OUTC; ++c) { float dv = rp[c] - mean; var = fmaf(dv, dv, var); }
    var *= (1.f / D_OUTC);
    float inv = rsqrtf(var + 1e-5f);
    size_t base = (size_t)(mTile * 16 + tid) * D_OUTC;
    for (int c = 0; c < D_OUTC; ++c)
      out[base + c] = (rp[c] - mean) * inv * ln_g[c] + ln_b[c];
  }
}

// ---------------- host launcher ----------------
extern "C" void kernel_launch(void* const* d_in, const int* in_sizes, int n_in,
                              void* d_out, int out_size, void* d_ws, size_t ws_size,
                              hipStream_t stream) {
  const float* node_feats = (const float*)d_in[0];
  const float* edge_feats = (const float*)d_in[1];
  const float* edge_dt    = (const float*)d_in[2];
  const float* time_w     = (const float*)d_in[3];
  const float* time_b     = (const float*)d_in[4];
  const float* wq_w = (const float*)d_in[5];
  const float* wq_b = (const float*)d_in[6];
  const float* wk_w = (const float*)d_in[7];
  const float* wk_b = (const float*)d_in[8];
  const float* wv_w = (const float*)d_in[9];
  const float* wv_b = (const float*)d_in[10];
  const float* wo_w = (const float*)d_in[11];
  const float* wo_b = (const float*)d_in[12];
  const float* ln_g = (const float*)d_in[13];
  const float* ln_b = (const float*)d_in[14];
  const int*   edge_dst = (const int*)d_in[15];
  float* out = (float*)d_out;

  char* ws = (char*)d_ws;
  __bf16* Bk = (__bf16*)(ws + BK_OFF);
  __bf16* Bv = (__bf16*)(ws + BV_OFF);
  __bf16* Bq = (__bf16*)(ws + BQ_OFF);
  __bf16* Bo = (__bf16*)(ws + BO_OFF);
  float* Qbuf  = (float*)(ws + Q_OFF);
  float* Kbuf  = (float*)(ws + K_OFF);
  float* Vbuf  = (float*)(ws + V_OFF);
  float* score = (float*)(ws + SC_OFF);
  float* aggb  = (float*)(ws + AG_OFF);

  prep_b_kernel<<<(KV_KP * NP + 255) / 256, 256, 0, stream>>>(
      wq_w, wk_w, wv_w, wo_w, Bq, Bk, Bv, Bo);

  q_gemm_kernel<<<NUM_DSTN / 16, 256, 0, stream>>>(
      node_feats, time_b, Bq, wq_b, Qbuf);

  kv_gemm_kernel<<<NUM_EDG / 16, 256, 0, stream>>>(
      node_feats, edge_feats, edge_dt, time_w, time_b, Bk, Bv, wk_b, wv_b,
      Kbuf, Vbuf);

  score_kernel<<<(NUM_EDG + 255) / 256, 256, 0, stream>>>(
      Qbuf, Kbuf, edge_dst, score);

  agg_kernel<<<NUM_DSTN / 8, 256, 0, stream>>>(Vbuf, score, edge_dst, aggb);

  out_gemm_ln_kernel<<<NUM_DSTN / 16, 256, 0, stream>>>(
      aggb, node_feats, Bo, wo_b, ln_g, ln_b, out);
}
